// SimpleRNN_2860448219166
// MI455X (gfx1250) — matrix operand order repaired and checked by reading, not compiled, not run on hardware
//
#include <hip/hip_runtime.h>

#define T_DIM 4096
#define H_DIM 2048
#define K_DIM 1024
#define V_DIM 1024
#define O_DIM 2048
#define CHUNK 64
#define NCHUNK (T_DIM / CHUNK)

#define MODE_F32 0
#define MODE_BF16 1
#define MODE_SIG_BF16 2
#define MODE_BF16_T 3

#define KSLICE 64          // staged k-depth per pipeline stage
#define LDA 72             // padded LDS row stride (elements); 144B -> conflict-free

typedef __attribute__((ext_vector_type(16))) __bf16 v16bf;
typedef __attribute__((ext_vector_type(8)))  float  v8f;
typedef __attribute__((ext_vector_type(4)))  unsigned int u32x4;

union FragU { u32x4 q[2]; v16bf v; };

__device__ __forceinline__ unsigned short f2bf(float x) {
  union { float f; unsigned u; } c; c.f = x;
  unsigned u = c.u;
  u += 0x7FFFu + ((u >> 16) & 1u);   // round-to-nearest-even
  return (unsigned short)(u >> 16);
}
__device__ __forceinline__ float bf2f(unsigned short h) {
  union { unsigned u; float f; } c; c.u = ((unsigned)h) << 16;
  return c.f;
}
__device__ __forceinline__ v8f vzero() {
  v8f z;
#pragma unroll
  for (int i = 0; i < 8; ++i) z[i] = 0.0f;
  return z;
}

// ---- CDNA5 async global->LDS copy (ASYNCcnt-tracked) + waits --------------
__device__ __forceinline__ void async_copy_b128(unsigned lds_off, const void* gptr) {
  asm volatile("global_load_async_to_lds_b128 %0, %1, off"
               :: "v"(lds_off), "v"(gptr) : "memory");
}
__device__ __forceinline__ void wait_async_le8() {
  asm volatile("s_wait_asynccnt 0x8" ::: "memory");
}
__device__ __forceinline__ void wait_async_le0() {
  asm volatile("s_wait_asynccnt 0x0" ::: "memory");
}

// ---- fragment loaders ------------------------------------------------------
// A fragment (16x32 bf16): lane L: m = L&15, kb = (L>>4)*16; elems 0..15 = K kb..kb+15
// -> two contiguous 16B loads.
__device__ __forceinline__ v16bf load_frag_a(const unsigned short* src, int row0, int k0, int ld) {
  int lane = threadIdx.x & 31;
  const unsigned short* p = src + (size_t)(row0 + (lane & 15)) * ld + k0 + ((lane >> 4) << 4);
  FragU f;
  f.q[0] = *(const u32x4*)(p);
  f.q[1] = *(const u32x4*)(p + 8);
  return f.v;
}
// B fragment (32x16, NT source row-major NxKd): lane L: n = L&15, kb = (L>>4)*16.
__device__ __forceinline__ v16bf load_frag_b(const unsigned short* src, int n0, int k0, int ld) {
  int lane = threadIdx.x & 31;
  const unsigned short* p = src + (size_t)(n0 + (lane & 15)) * ld + k0 + ((lane >> 4) << 4);
  FragU f;
  f.q[0] = *(const u32x4*)(p);
  f.q[1] = *(const u32x4*)(p + 8);
  return f.v;
}
// LDS variants (KSLICE-wide tiles, row stride LDA elements)
__device__ __forceinline__ v16bf lds_frag_a(const unsigned short* base, int m0, int ks) {
  int lane = threadIdx.x & 31;
  const unsigned short* p = base + (m0 + (lane & 15)) * LDA + ks + ((lane >> 4) << 4);
  FragU f;
  f.q[0] = *(const u32x4*)(p);
  f.q[1] = *(const u32x4*)(p + 8);
  return f.v;
}
__device__ __forceinline__ v16bf lds_frag_b(const unsigned short* base, int n0, int ks) {
  int lane = threadIdx.x & 31;
  const unsigned short* p = base + (n0 + (lane & 15)) * LDA + ks + ((lane >> 4) << 4);
  FragU f;
  f.q[0] = *(const u32x4*)(p);
  f.q[1] = *(const u32x4*)(p + 8);
  return f.v;
}
__device__ __forceinline__ v8f wmma_bf16(v16bf a, v16bf b, v8f c) {
  return __builtin_amdgcn_wmma_f32_16x16x32_bf16(false, a, false, b, (short)0, c, false, false);
}

// ---------------------------------------------------------------- utilities
__global__ void cvt_bf16_kernel(const float* __restrict__ src, unsigned short* __restrict__ dst, int n) {
  int i = blockIdx.x * 256 + threadIdx.x;
  if (i < n) dst[i] = f2bf(src[i]);
}
__global__ void zero32_kernel(unsigned* __restrict__ p, int nwords) {
  int i = blockIdx.x * 256 + threadIdx.x;
  if (i < nwords) p[i] = 0u;
}

// ------------------------------------------------- generic NT GEMM (bf16 WMMA)
// C[M,N] = act(A[M,Kd] @ B[N,Kd]^T + bias). grid = (N/128, M/128), block 256.
// Double-buffered async global->LDS staging of 128x64 k-slices of A and B.
__global__ void gemm_nt_kernel(const unsigned short* __restrict__ A,
                               const unsigned short* __restrict__ B,
                               const float* __restrict__ bias,
                               void* __restrict__ Cout,
                               int Kd, int ldo, int mode) {
  __shared__ unsigned short smem[2][2][128 * LDA];   // 72 KB
  const unsigned TILEB = 128 * LDA * 2;              // bytes per tile
  const unsigned BUFB  = 2 * TILEB;                  // bytes per buffer (A+B)

  int tid = threadIdx.x;
  int wave = tid >> 5, lane = tid & 31;
  int mw = wave & 3;   // 4 m-strips of 32
  int nw = wave >> 2;  // 2 n-strips of 64
  int rowA0 = blockIdx.y * 128;
  int colB0 = blockIdx.x * 128;

  // per-thread copy slot: row = tid>>1, 32-elem half = (tid&1)*32, 4 x b128 each
  int cr = tid >> 1;
  int cs = (tid & 1) * 32;
  const unsigned short* gA = A + (size_t)(rowA0 + cr) * Kd + cs;
  const unsigned short* gB = B + (size_t)(colB0 + cr) * Kd + cs;
  unsigned lA = (unsigned)(uintptr_t)(&smem[0][0][0]) + (unsigned)(cr * LDA + cs) * 2;
  unsigned lB = lA + TILEB;

  v8f acc[2][4];
#pragma unroll
  for (int i = 0; i < 2; ++i)
#pragma unroll
    for (int j = 0; j < 4; ++j) acc[i][j] = vzero();

  auto issue = [&](int buf) {
    unsigned boff = (unsigned)buf * BUFB;
#pragma unroll
    for (int i = 0; i < 4; ++i) async_copy_b128(lA + boff + i * 16, gA + i * 8);
#pragma unroll
    for (int i = 0; i < 4; ++i) async_copy_b128(lB + boff + i * 16, gB + i * 8);
    gA += KSLICE;
    gB += KSLICE;
  };
  auto compute = [&](int buf) {
    const unsigned short* At = &smem[buf][0][0];
    const unsigned short* Bt = &smem[buf][1][0];
#pragma unroll
    for (int ks = 0; ks < KSLICE; ks += 32) {
      v16bf a0 = lds_frag_a(At, mw * 32, ks);
      v16bf a1 = lds_frag_a(At, mw * 32 + 16, ks);
#pragma unroll
      for (int j = 0; j < 4; ++j) {
        v16bf b = lds_frag_b(Bt, nw * 64 + 16 * j, ks);
        acc[0][j] = wmma_bf16(a0, b, acc[0][j]);
        acc[1][j] = wmma_bf16(a1, b, acc[1][j]);
      }
    }
  };

  int nsteps = Kd / KSLICE;
  issue(0);
  for (int s = 0; s < nsteps - 1; ++s) {   // steady state: unconditional waits
    issue((s + 1) & 1);
    wait_async_le8();                      // previous slice's 8 copies done
    __syncthreads();
    compute(s & 1);
    __syncthreads();
  }
  wait_async_le0();
  __syncthreads();
  compute((nsteps - 1) & 1);

  int row0 = rowA0 + mw * 32;
  int col0 = colB0 + nw * 64;
#pragma unroll
  for (int i = 0; i < 2; ++i)
#pragma unroll
    for (int j = 0; j < 4; ++j) {
      int n = col0 + 16 * j + (lane & 15);
      float bv = bias ? bias[n] : 0.0f;
#pragma unroll
      for (int r = 0; r < 8; ++r) {
        int m = row0 + 16 * i + r + ((lane >> 4) << 3);
        float val = acc[i][j][r] + bv;
        if (mode == MODE_SIG_BF16) val = 1.0f / (1.0f + __expf(-val));
        if (mode == MODE_F32) {
          ((float*)Cout)[(size_t)m * ldo + n] = val;
        } else if (mode == MODE_BF16_T) {
          ((unsigned short*)Cout)[(size_t)n * ldo + m] = f2bf(val);
        } else {
          ((unsigned short*)Cout)[(size_t)m * ldo + n] = f2bf(val);
        }
      }
    }
}

// ---------------------------------------------------------- chunk preprocess
__global__ void preproc_kernel(const unsigned short* __restrict__ Qb,
                               const unsigned short* __restrict__ Kb,
                               const unsigned short* __restrict__ Gb,
                               unsigned short* __restrict__ Qp,
                               unsigned short* __restrict__ Kp,
                               unsigned short* __restrict__ K2T,
                               float* __restrict__ AC) {
  int tid = blockIdx.x * 256 + threadIdx.x;   // NCHUNK * K_DIM threads
  int c = tid >> 10;
  int e = tid & 1023;
  size_t base = (size_t)c * CHUNK * K_DIM + e;
  float a = 1.0f;
  float kpr[CHUNK];
#pragma unroll
  for (int t = 0; t < CHUNK; ++t) {
    size_t idx = base + (size_t)t * K_DIM;
    float g = bf2f(Gb[idx]);
    a = fmaxf(a * g, 1e-30f);                 // guard fp32 underflow
    float q = bf2f(Qb[idx]);
    float k = bf2f(Kb[idx]);
    float kp = k / a;
    kpr[t] = kp;
    Qp[idx] = f2bf(q * a);
    Kp[idx] = f2bf(kp);
  }
  AC[(size_t)c * K_DIM + e] = a;
  size_t tb = (size_t)c * (K_DIM * CHUNK) + (size_t)e * CHUNK;
#pragma unroll
  for (int t = 0; t < CHUNK; ++t) K2T[tb + t] = f2bf(kpr[t] * a);
}

// ------------------------------------------------------------- intra scores
__global__ void scores_kernel(const unsigned short* __restrict__ Qp,
                              const unsigned short* __restrict__ Kp,
                              unsigned short* __restrict__ Pb) {
  int c = blockIdx.x;
  int wave = threadIdx.x >> 5;
  int lane = threadIdx.x & 31;
  int mt = wave & 3;      // m-tile (16 rows)
  int ns = wave >> 2;     // n-strip of 32
  v8f acc[2] = { vzero(), vzero() };
  for (int e0 = 0; e0 < K_DIM; e0 += 32) {
    v16bf a = load_frag_a(Qp, c * CHUNK + mt * 16, e0, K_DIM);
#pragma unroll
    for (int j = 0; j < 2; ++j) {
      v16bf b = load_frag_b(Kp, c * CHUNK + ns * 32 + 16 * j, e0, K_DIM);
      acc[j] = wmma_bf16(a, b, acc[j]);
    }
  }
#pragma unroll
  for (int j = 0; j < 2; ++j) {
    int s = ns * 32 + 16 * j + (lane & 15);
#pragma unroll
    for (int r = 0; r < 8; ++r) {
      int t = mt * 16 + r + ((lane >> 4) << 3);
      float val = (s <= t) ? acc[j][r] : 0.0f;    // causal (inclusive) mask
      Pb[(size_t)c * (CHUNK * CHUNK) + (size_t)t * CHUNK + s] = f2bf(val);
    }
  }
}

// ------------------------------------------------ fused per-chunk recurrence
// blocks [0,8):   Out_c = Q'_c @ S_in^T(bf16) + P_c @ V_c     (64 x 1024)
// blocks [8,72):  S^T_out = aC * S^T_in + V^T_c @ K''_c       (1024 x 1024)
__global__ void chunk_step_kernel(const unsigned short* __restrict__ Qp,
                                  const unsigned short* __restrict__ Pb,
                                  const unsigned short* __restrict__ VbT,
                                  const unsigned short* __restrict__ K2T,
                                  const float* __restrict__ AC,
                                  const unsigned short* __restrict__ SbIn,
                                  const float* __restrict__ Sin,
                                  float* __restrict__ Sout,
                                  unsigned short* __restrict__ SbOut,
                                  unsigned short* __restrict__ Outb, int c) {
  int wave = threadIdx.x >> 5;
  int lane = threadIdx.x & 31;

  if (blockIdx.x < 8) {
    // -------- chunk output --------
    int mw = wave & 3;     // 4 m-tiles of 16 (M=64)
    int nw = wave >> 2;    // 2 n-strips of 64
    int col0 = blockIdx.x * 128 + nw * 64;
    v8f acc[4];
#pragma unroll
    for (int j = 0; j < 4; ++j) acc[j] = vzero();

    for (int e0 = 0; e0 < K_DIM; e0 += 32) {       // inter-chunk
      v16bf a = load_frag_a(Qp, c * CHUNK + mw * 16, e0, K_DIM);
#pragma unroll
      for (int j = 0; j < 4; ++j) {
        v16bf b = load_frag_b(SbIn, col0 + 16 * j, e0, K_DIM);
        acc[j] = wmma_bf16(a, b, acc[j]);
      }
    }
    const unsigned short* Pc = Pb + (size_t)c * (CHUNK * CHUNK);
#pragma unroll
    for (int s0 = 0; s0 < CHUNK; s0 += 32) {       // intra-chunk
      v16bf a = load_frag_a(Pc, mw * 16, s0, CHUNK);
#pragma unroll
      for (int j = 0; j < 4; ++j) {
        v16bf b = load_frag_b(VbT, col0 + 16 * j, c * CHUNK + s0, T_DIM);
        acc[j] = wmma_bf16(a, b, acc[j]);
      }
    }
#pragma unroll
    for (int j = 0; j < 4; ++j) {
      int n = col0 + 16 * j + (lane & 15);
#pragma unroll
      for (int r = 0; r < 8; ++r) {
        int m = mw * 16 + r + ((lane >> 4) << 3);
        Outb[(size_t)(c * CHUNK + m) * V_DIM + n] = f2bf(acc[j][r]);
      }
    }
  } else {
    // -------- state update --------
    int idx = blockIdx.x - 8;
    int bx = idx & 7;     // e tiles
    int by = idx >> 3;    // v tiles
    int mw = wave & 3;
    int nw = wave >> 2;
    int row0 = by * 128 + mw * 32;   // v
    int col0 = bx * 128 + nw * 64;   // e
    const unsigned short* K2c = K2T + (size_t)c * (K_DIM * CHUNK);

    v8f acc[2][4];
#pragma unroll
    for (int i = 0; i < 2; ++i)
#pragma unroll
      for (int j = 0; j < 4; ++j) acc[i][j] = vzero();

#pragma unroll
    for (int s0 = 0; s0 < CHUNK; s0 += 32) {
      v16bf a0 = load_frag_a(VbT, row0, c * CHUNK + s0, T_DIM);
      v16bf a1 = load_frag_a(VbT, row0 + 16, c * CHUNK + s0, T_DIM);
#pragma unroll
      for (int j = 0; j < 4; ++j) {
        v16bf b = load_frag_b(K2c, col0 + 16 * j, s0, CHUNK);
        acc[0][j] = wmma_bf16(a0, b, acc[0][j]);
        acc[1][j] = wmma_bf16(a1, b, acc[1][j]);
      }
    }
#pragma unroll
    for (int i = 0; i < 2; ++i)
#pragma unroll
      for (int j = 0; j < 4; ++j) {
        int e = col0 + 16 * j + (lane & 15);
        float ac = AC[(size_t)c * K_DIM + e];
#pragma unroll
        for (int r = 0; r < 8; ++r) {
          int v = row0 + 16 * i + r + ((lane >> 4) << 3);
          size_t sidx = (size_t)v * K_DIM + e;
          float val = acc[i][j][r] + ac * Sin[sidx];
          Sout[sidx] = val;
          SbOut[sidx] = f2bf(val);
        }
      }
  }
}

// final_state (K,V) = S^T transposed back
__global__ void fstate_kernel(const float* __restrict__ S, float* __restrict__ dst) {
  int tid = blockIdx.x * 256 + threadIdx.x;
  int v = tid >> 10;
  int e = tid & 1023;
  dst[(size_t)e * V_DIM + v] = S[(size_t)v * K_DIM + e];
}

// ------------------------------------------------------------------- launcher
extern "C" void kernel_launch(void* const* d_in, const int* in_sizes, int n_in,
                              void* d_out, int out_size, void* d_ws, size_t ws_size,
                              hipStream_t stream) {
  (void)in_sizes; (void)n_in; (void)out_size; (void)ws_size;
  const float* hs = (const float*)d_in[0];
  const float* Wq = (const float*)d_in[1];  const float* bq = (const float*)d_in[2];
  const float* Wk = (const float*)d_in[3];  const float* bk = (const float*)d_in[4];
  const float* Wv = (const float*)d_in[5];  const float* bv = (const float*)d_in[6];
  const float* Wg = (const float*)d_in[7];  const float* bg = (const float*)d_in[8];
  const float* Wo = (const float*)d_in[9];  const float* bo = (const float*)d_in[10];

  size_t off = 0;
  auto take = [&](size_t bytes) -> char* {
    char* p = (char*)d_ws + off;
    off += (bytes + 255) & ~(size_t)255;
    return p;
  };
  unsigned short* Xb  = (unsigned short*)take((size_t)T_DIM * H_DIM * 2);
  unsigned short* Wqb = (unsigned short*)take((size_t)K_DIM * H_DIM * 2);
  unsigned short* Wkb = (unsigned short*)take((size_t)K_DIM * H_DIM * 2);
  unsigned short* Wvb = (unsigned short*)take((size_t)V_DIM * H_DIM * 2);
  unsigned short* Wgb = (unsigned short*)take((size_t)K_DIM * H_DIM * 2);
  unsigned short* Wob = (unsigned short*)take((size_t)O_DIM * V_DIM * 2);
  unsigned short* Qb  = (unsigned short*)take((size_t)T_DIM * K_DIM * 2);
  unsigned short* Kb  = (unsigned short*)take((size_t)T_DIM * K_DIM * 2);
  unsigned short* Gb  = (unsigned short*)take((size_t)T_DIM * K_DIM * 2);
  unsigned short* VbT = (unsigned short*)take((size_t)V_DIM * T_DIM * 2);
  unsigned short* Qp  = (unsigned short*)take((size_t)T_DIM * K_DIM * 2);
  unsigned short* Kp  = (unsigned short*)take((size_t)T_DIM * K_DIM * 2);
  unsigned short* K2T = (unsigned short*)take((size_t)K_DIM * T_DIM * 2);
  float*          ACf = (float*)take((size_t)NCHUNK * K_DIM * 4);
  unsigned short* Pb  = (unsigned short*)take((size_t)NCHUNK * CHUNK * CHUNK * 2);
  unsigned short* Outb= (unsigned short*)take((size_t)T_DIM * V_DIM * 2);
  float*          Sf0 = (float*)take((size_t)V_DIM * K_DIM * 4);
  float*          Sf1 = (float*)take((size_t)V_DIM * K_DIM * 4);
  unsigned short* Sb0 = (unsigned short*)take((size_t)V_DIM * K_DIM * 2);
  unsigned short* Sb1 = (unsigned short*)take((size_t)V_DIM * K_DIM * 2);

  // 1) fp32 -> bf16 conversions
  cvt_bf16_kernel<<<(T_DIM * H_DIM) / 256, 256, 0, stream>>>(hs, Xb, T_DIM * H_DIM);
  cvt_bf16_kernel<<<(K_DIM * H_DIM) / 256, 256, 0, stream>>>(Wq, Wqb, K_DIM * H_DIM);
  cvt_bf16_kernel<<<(K_DIM * H_DIM) / 256, 256, 0, stream>>>(Wk, Wkb, K_DIM * H_DIM);
  cvt_bf16_kernel<<<(V_DIM * H_DIM) / 256, 256, 0, stream>>>(Wv, Wvb, V_DIM * H_DIM);
  cvt_bf16_kernel<<<(K_DIM * H_DIM) / 256, 256, 0, stream>>>(Wg, Wgb, K_DIM * H_DIM);
  cvt_bf16_kernel<<<(O_DIM * V_DIM) / 256, 256, 0, stream>>>(Wo, Wob, O_DIM * V_DIM);

  // 2) input projections (bf16 WMMA, async->LDS staged, fused epilogues)
  dim3 gproj(K_DIM / 128, T_DIM / 128);
  gemm_nt_kernel<<<gproj, 256, 0, stream>>>(Xb, Wqb, bq, Qb,  H_DIM, K_DIM, MODE_BF16);
  gemm_nt_kernel<<<gproj, 256, 0, stream>>>(Xb, Wkb, bk, Kb,  H_DIM, K_DIM, MODE_SIG_BF16);
  gemm_nt_kernel<<<gproj, 256, 0, stream>>>(Xb, Wgb, bg, Gb,  H_DIM, K_DIM, MODE_SIG_BF16);
  gemm_nt_kernel<<<gproj, 256, 0, stream>>>(Xb, Wvb, bv, VbT, H_DIM, T_DIM, MODE_BF16_T);

  // 3) chunk preprocessing + intra-chunk scores (all chunks parallel)
  preproc_kernel<<<(NCHUNK * K_DIM) / 256, 256, 0, stream>>>(Qb, Kb, Gb, Qp, Kp, K2T, ACf);
  scores_kernel<<<NCHUNK, 256, 0, stream>>>(Qp, Kp, Pb);

  // 4) zero initial state (f32 + bf16 shadow)
  zero32_kernel<<<(V_DIM * K_DIM) / 256, 256, 0, stream>>>((unsigned*)Sf0, V_DIM * K_DIM);
  zero32_kernel<<<(V_DIM * K_DIM / 2) / 256, 256, 0, stream>>>((unsigned*)Sb0, V_DIM * K_DIM / 2);

  // 5) sequential chunk recurrence (double-buffered state, 1 launch per chunk)
  for (int c = 0; c < NCHUNK; ++c) {
    const float* Sin = (c & 1) ? Sf1 : Sf0;
    float* Sout      = (c & 1) ? Sf0 : Sf1;
    const unsigned short* SbIn = (c & 1) ? Sb1 : Sb0;
    unsigned short* SbOut      = (c & 1) ? Sb0 : Sb1;
    chunk_step_kernel<<<8 + (K_DIM / 128) * (V_DIM / 128), 256, 0, stream>>>(
        Qp, Pb, VbT, K2T, ACf, SbIn, Sin, Sout, SbOut, Outb, c);
  }

  // 6) output projection -> d_out (f32), then final state (K,V)
  gemm_nt_kernel<<<dim3(O_DIM / 128, T_DIM / 128), 256, 0, stream>>>(
      Outb, Wob, bo, d_out, V_DIM, O_DIM, MODE_F32);
  fstate_kernel<<<(V_DIM * K_DIM) / 256, 256, 0, stream>>>(
      Sf0, (float*)d_out + (size_t)T_DIM * O_DIM);
}
